// AttentionBlock_54571854463422
// MI455X (gfx1250) — matrix operand order repaired and checked by reading, not compiled, not run on hardware
//
#include <hip/hip_runtime.h>
#include <hip/hip_bf16.h>

typedef __attribute__((ext_vector_type(16))) _Float16 v16h;
typedef __attribute__((ext_vector_type(8)))  _Float16 v8h;
typedef __attribute__((ext_vector_type(8)))  float    v8f;

#define BB   4
#define CC   256
#define NN   4096
#define RDIM 32

static __device__ __forceinline__ v16h make_frag(v8h lo, v8h hi) {
  v16h f;
#pragma unroll
  for (int i = 0; i < 8; ++i) { f[i] = lo[i]; f[i + 8] = hi[i]; }
  return f;
}

// A-fragment (16 x K32, f16), row-major source, leading dim ld (f16 elems).
// Lane L holds row M=(L&15); K chunks at (L>>4)*8 and (L>>4)*8+16.
static __device__ __forceinline__ v16h load_fragA(const _Float16* base, int ld) {
  const int lane = threadIdx.x & 31;
  const _Float16* p = base + (size_t)(lane & 15) * ld + (lane >> 4) * 8;
  v8h lo = *(const v8h*)(p);
  v8h hi = *(const v8h*)(p + 16);
  return make_frag(lo, hi);
}

// B-fragment (K32 x 16, f16) from "N-major" memory (each N row holds contiguous K).
// Lane L holds column N=(L&15); K chunks at (L>>4)*8 and (L>>4)*8+16.
static __device__ __forceinline__ v16h load_fragB(const _Float16* base, int ld) {
  const int lane = threadIdx.x & 31;
  const _Float16* p = base + (size_t)(lane & 15) * ld + (lane >> 4) * 8;
  v8h lo = *(const v8h*)(p);
  v8h hi = *(const v8h*)(p + 16);
  return make_frag(lo, hi);
}

static __device__ __forceinline__ v8f wmma_f16(v16h a, v16h b, v8f c) {
  return __builtin_amdgcn_wmma_f32_16x16x32_f16(false, a, false, b, (short)0, c,
                                                false, false);
}

// ---------------------------------------------------------------------------
// K0: transpose-convert x [B][C][N] f32  ->  xhT [B][N][C] f16
// ---------------------------------------------------------------------------
__global__ __launch_bounds__(256) void xpose_kernel(const float* __restrict__ x,
                                                    _Float16* __restrict__ xhT) {
  __shared__ float tile[32][33];
  const int tx = threadIdx.x & 31, ty = threadIdx.x >> 5;
  const int b = blockIdx.z, c0 = blockIdx.y * 32, n0 = blockIdx.x * 32;
#pragma unroll
  for (int i = 0; i < 4; ++i)
    tile[ty + i * 8][tx] =
        x[((size_t)b * CC + c0 + ty + i * 8) * NN + n0 + tx];
  __syncthreads();
#pragma unroll
  for (int i = 0; i < 4; ++i)
    xhT[((size_t)b * NN + n0 + ty + i * 8) * CC + c0 + tx] =
        (_Float16)tile[tx][ty + i * 8];
}

// ---------------------------------------------------------------------------
// K0b: Wv f32 -> f16
// ---------------------------------------------------------------------------
__global__ __launch_bounds__(256) void wvcvt_kernel(const float* __restrict__ Wv,
                                                    _Float16* __restrict__ Wvh) {
  int i = blockIdx.x * 256 + threadIdx.x;
  if (i < CC * CC) Wvh[i] = (_Float16)Wv[i];
}

// ---------------------------------------------------------------------------
// K1: q/k projections (tiny GEMMs, scalar f32) -> qh/kh [B][N][RD] f16
// ---------------------------------------------------------------------------
__global__ __launch_bounds__(256) void qkproj_kernel(
    const float* __restrict__ x, const float* __restrict__ Wq,
    const float* __restrict__ bq, const float* __restrict__ Wk,
    const float* __restrict__ bk, _Float16* __restrict__ qh,
    _Float16* __restrict__ kh) {
  const int t = threadIdx.x;
  const int r = t & 31;
  const int nl = t >> 5;
  const int b = blockIdx.x / (NN / 8);
  const int n = (blockIdx.x % (NN / 8)) * 8 + nl;
  const float* xp = x + (size_t)b * CC * NN + n;
  float aq = bq[r], ak = bk[r];
#pragma unroll 4
  for (int c = 0; c < CC; ++c) {
    float xv = xp[(size_t)c * NN];
    aq += Wq[r * CC + c] * xv;
    ak += Wk[r * CC + c] * xv;
  }
  size_t o = ((size_t)b * NN + n) * RDIM + r;
  qh[o] = (_Float16)aq;
  kh[o] = (_Float16)ak;
}

// ---------------------------------------------------------------------------
// K2: v = Wv @ x + bv via WMMA -> vh [B][C][N] f16.
// Two n-tiles per wave: each Wv A-fragment feeds two independent WMMAs
// (amortizing its load latency), and A traffic is halved.  Next-kk fragment
// loads are issued before the current WMMAs (software pipeline).
// grid (N/32, B), 4 waves; wave w owns c-chunk [w*64, w*64+64)
// ---------------------------------------------------------------------------
__global__ __launch_bounds__(128) void vproj_kernel(
    const _Float16* __restrict__ Wvh, const _Float16* __restrict__ xhT,
    const float* __restrict__ bv, _Float16* __restrict__ vh) {
  const int tid = threadIdx.x;
  const int w = tid >> 5, lane = tid & 31;
  const int h = lane >> 4, li = lane & 15;
  const int b = blockIdx.y, n0 = blockIdx.x * 32;

  v8f acc[4][2];
#pragma unroll
  for (int mt = 0; mt < 4; ++mt) { acc[mt][0] = v8f{}; acc[mt][1] = v8f{}; }

  const _Float16* xrow0 = xhT + ((size_t)b * NN + n0) * CC;
  const _Float16* xrow1 = xrow0 + (size_t)16 * CC;
  const _Float16* wrow = Wvh + (size_t)(w * 64) * CC;

  v16h bX0 = load_fragB(xrow0, CC);
  v16h bX1 = load_fragB(xrow1, CC);
  v16h aW[4], aWn[4];
#pragma unroll
  for (int mt = 0; mt < 4; ++mt)
    aW[mt] = load_fragA(wrow + (size_t)(mt * 16) * CC, CC);

#pragma unroll
  for (int kk = 0; kk < CC / 32; ++kk) {
    v16h bX0n, bX1n;
    if (kk + 1 < CC / 32) {                       // issue next-iter loads first
      bX0n = load_fragB(xrow0 + (kk + 1) * 32, CC);
      bX1n = load_fragB(xrow1 + (kk + 1) * 32, CC);
#pragma unroll
      for (int mt = 0; mt < 4; ++mt)
        aWn[mt] = load_fragA(wrow + (size_t)(mt * 16) * CC + (kk + 1) * 32, CC);
    }
#pragma unroll
    for (int mt = 0; mt < 4; ++mt) {
      acc[mt][0] = wmma_f16(aW[mt], bX0, acc[mt][0]);
      acc[mt][1] = wmma_f16(aW[mt], bX1, acc[mt][1]);
    }
    if (kk + 1 < CC / 32) {
      bX0 = bX0n; bX1 = bX1n;
#pragma unroll
      for (int mt = 0; mt < 4; ++mt) aW[mt] = aWn[mt];
    }
  }
#pragma unroll
  for (int mt = 0; mt < 4; ++mt)
#pragma unroll
    for (int g = 0; g < 8; ++g) {
      int c = w * 64 + mt * 16 + g + 8 * h;
      float bias = bv[c];
      size_t rowo = ((size_t)b * CC + c) * NN + n0 + li;
      vh[rowo] = (_Float16)(acc[mt][0][g] + bias);
      vh[rowo + 16] = (_Float16)(acc[mt][1][g] + bias);
    }
}

// ---------------------------------------------------------------------------
// K3: fused flash attention + residual.
// 256 threads (8 waves) per block; block owns 64 query rows (4 i-tiles) and
// all 256 channels (each wave owns a 32-channel O^T slice).
// Waves 0..3 each run the online softmax for ONE i-tile (no duplicated exps),
// staging P in double-buffered LDS; ONE barrier per 64-j block hands P to all
// waves.  Row-sums of P are computed with a WMMA against a ones B-fragment.
// ---------------------------------------------------------------------------
__global__ __launch_bounds__(256) void attn_kernel(
    const _Float16* __restrict__ qh, const _Float16* __restrict__ kh,
    const _Float16* __restrict__ vh, const float* __restrict__ x,
    const float* __restrict__ gamma, float* __restrict__ out) {
  __shared__ __align__(32) _Float16 sP[2][4][16 * 64];  // [buf][i-tile][i][j]
  __shared__ float sA[2][4][16];                        // alpha broadcast
  __shared__ float sL[4][16];                           // final row sums

  const int tid = threadIdx.x;
  const int w = tid >> 5, lane = tid & 31;
  const int h = lane >> 4, li = lane & 15;
  const int b = blockIdx.y, i0 = blockIdx.x * 64;

  v16h aQ;
  if (w < 4) aQ = load_fragA(qh + ((size_t)b * NN + i0 + w * 16) * RDIM, RDIM);

  v16h vone;
#pragma unroll
  for (int i = 0; i < 16; ++i) vone[i] = (_Float16)1.0f;

  v8f O[4][2];  // [i-tile][c-subtile]
#pragma unroll
  for (int it = 0; it < 4; ++it) { O[it][0] = v8f{}; O[it][1] = v8f{}; }
  float m[8], l[8];
#pragma unroll
  for (int g = 0; g < 8; ++g) { m[g] = -1e30f; l[g] = 0.0f; }

  const _Float16* vbase = vh + ((size_t)b * CC + w * 32) * NN;

  int buf = 0;
  for (int j0 = 0; j0 < NN; j0 += 64, buf ^= 1) {
    if (w < 4) {
      const _Float16* kb = kh + ((size_t)b * NN + j0) * RDIM;
      v16h bK0 = load_fragB(kb, RDIM);
      v16h bK1 = load_fragB(kb + 16 * RDIM, RDIM);
      v16h bK2 = load_fragB(kb + 32 * RDIM, RDIM);
      v16h bK3 = load_fragB(kb + 48 * RDIM, RDIM);
      if (j0 + 64 < NN) __builtin_prefetch(kb + 64 * RDIM, 0, 3);
      v8f z = {};
      v8f S0 = wmma_f16(aQ, bK0, z);
      v8f S1 = wmma_f16(aQ, bK1, z);
      v8f S2 = wmma_f16(aQ, bK2, z);
      v8f S3 = wmma_f16(aQ, bK3, z);

      float alpha[8];
      _Float16* Pw = &sP[buf][w][0];
#pragma unroll
      for (int g = 0; g < 8; ++g) {
        float bmax = fmaxf(fmaxf(S0[g], S1[g]), fmaxf(S2[g], S3[g]));
        bmax = fmaxf(bmax, __shfl_xor(bmax, 1));
        bmax = fmaxf(bmax, __shfl_xor(bmax, 2));
        bmax = fmaxf(bmax, __shfl_xor(bmax, 4));
        bmax = fmaxf(bmax, __shfl_xor(bmax, 8));
        float mn = fmaxf(m[g], bmax);
        alpha[g] = __expf(m[g] - mn);
        m[g] = mn;
        int row = (g + 8 * h) * 64 + li;
        Pw[row +  0] = (_Float16)__expf(S0[g] - mn);
        Pw[row + 16] = (_Float16)__expf(S1[g] - mn);
        Pw[row + 32] = (_Float16)__expf(S2[g] - mn);
        Pw[row + 48] = (_Float16)__expf(S3[g] - mn);
        if (li == g) sA[buf][w][g + 8 * h] = alpha[g];
      }
      // row sums of P via WMMA against ones (lands in m/l striping)
      v16h aP0 = load_fragA(Pw, 64);
      v16h aP1 = load_fragA(Pw + 32, 64);
      v8f z2 = {};
      v8f Ssum = wmma_f16(aP0, vone, z2);
      Ssum = wmma_f16(aP1, vone, Ssum);
#pragma unroll
      for (int g = 0; g < 8; ++g) l[g] = l[g] * alpha[g] + Ssum[g];
    }

    // V fragments for this j-block: issue BEFORE the barrier so the loads
    // overlap barrier latency and the softmax tail.
    v16h aV00 = load_fragA(vbase + j0, NN);
    v16h aV10 = load_fragA(vbase + (size_t)16 * NN + j0, NN);
    v16h aV01 = load_fragA(vbase + j0 + 32, NN);
    v16h aV11 = load_fragA(vbase + (size_t)16 * NN + j0 + 32, NN);
    if (j0 + 64 < NN) __builtin_prefetch(vbase + j0 + 64, 0, 3);

    __syncthreads();

    // rescale O^T accumulators: column i of tile it -> alpha
#pragma unroll
    for (int it = 0; it < 4; ++it) {
      float av = sA[buf][it][li];
#pragma unroll
      for (int g = 0; g < 8; ++g) { O[it][0][g] *= av; O[it][1][g] *= av; }
    }
    // O^T += V x P^T for both 32-wide j sub-blocks
#pragma unroll
    for (int it = 0; it < 4; ++it) {
      v16h bP = load_fragB(&sP[buf][it][0], 64);
      O[it][0] = wmma_f16(aV00, bP, O[it][0]);
      O[it][1] = wmma_f16(aV10, bP, O[it][1]);
    }
#pragma unroll
    for (int it = 0; it < 4; ++it) {
      v16h bP = load_fragB(&sP[buf][it][32], 64);
      O[it][0] = wmma_f16(aV01, bP, O[it][0]);
      O[it][1] = wmma_f16(aV11, bP, O[it][1]);
    }
  }

  if (w < 4) {
#pragma unroll
    for (int g = 0; g < 8; ++g)
      if (li == g) sL[w][g + 8 * h] = l[g];
  }
  __syncthreads();

  const float gm = gamma[0];
#pragma unroll
  for (int it = 0; it < 4; ++it) {
    float linv = 1.0f / sL[it][li];
#pragma unroll
    for (int mt = 0; mt < 2; ++mt)
#pragma unroll
      for (int g = 0; g < 8; ++g) {
        int c = w * 32 + mt * 16 + g + 8 * h;
        size_t idx = ((size_t)b * CC + c) * NN + i0 + it * 16 + li;
        out[idx] = gm * (O[it][mt][g] * linv) + x[idx];
      }
  }
}

// ---------------------------------------------------------------------------
extern "C" void kernel_launch(void* const* d_in, const int* in_sizes, int n_in,
                              void* d_out, int out_size, void* d_ws,
                              size_t ws_size, hipStream_t stream) {
  (void)in_sizes; (void)n_in; (void)out_size; (void)ws_size;
  const float* x     = (const float*)d_in[0];
  const float* Wq    = (const float*)d_in[1];
  const float* bq    = (const float*)d_in[2];
  const float* Wk    = (const float*)d_in[3];
  const float* bk    = (const float*)d_in[4];
  const float* Wv    = (const float*)d_in[5];
  const float* bv    = (const float*)d_in[6];
  const float* gamma = (const float*)d_in[7];
  float* out = (float*)d_out;

  char* ws = (char*)d_ws;
  _Float16* xhT = (_Float16*)(ws);                  // 8 MB  [B][N][C]
  _Float16* vh  = (_Float16*)(ws + 8388608);        // 8 MB  [B][C][N]
  _Float16* qh  = (_Float16*)(ws + 16777216);       // 1 MB  [B][N][RD]
  _Float16* kh  = (_Float16*)(ws + 17825792);       // 1 MB  [B][N][RD]
  _Float16* Wvh = (_Float16*)(ws + 18874368);       // 128 KB [C][C]

  xpose_kernel<<<dim3(NN / 32, CC / 32, BB), 256, 0, stream>>>(x, xhT);
  wvcvt_kernel<<<(CC * CC + 255) / 256, 256, 0, stream>>>(Wv, Wvh);
  qkproj_kernel<<<BB * (NN / 8), 256, 0, stream>>>(x, Wq, bq, Wk, bk, qh, kh);
  vproj_kernel<<<dim3(NN / 32, BB), 128, 0, stream>>>(Wvh, xhT, bv, vh);
  attn_kernel<<<dim3(NN / 64, BB), 256, 0, stream>>>(qh, kh, vh, x, gamma, out);
}
